// SPROUTLayer_32865089749383
// MI455X (gfx1250) — matrix operand order repaired and checked by reading, not compiled, not run on hardware
//
#include <hip/hip_runtime.h>
#include <math.h>

#define Bsz 2
#define Ssz 1024
#define Dsz 512
#define Nsz 32
#define Hsz 2048
#define Ksel 4
#define NHsz 8
#define HDsz 64
#define TOK (Bsz * Ssz)   // 2048 tokens

typedef __attribute__((ext_vector_type(16))) _Float16 v16h;
typedef __attribute__((ext_vector_type(8)))  _Float16 v8h;
typedef __attribute__((ext_vector_type(8)))  float    v8f;

// ---------------------------------------------------------------- WMMA helpers
__device__ __forceinline__ v8f wmma16(v16h a, v16h b, v8f c) {
  // D = A(16x32 f16) * B(32x16 f16) + C(16x16 f32)
  return __builtin_amdgcn_wmma_f32_16x16x32_f16(false, a, false, b, (short)0, c,
                                                false, false);
}

// A fragment: 16x32 f16 tile from row-major storage (LDS or global).
// Lane layout: row = lane&15; g = lane>>4; K = g*16 + e (contiguous, matches frag_bT).
// Requires 16B alignment of (row*ld + k0 + g*16) in f16 units.
__device__ __forceinline__ v16h frag_a(const _Float16* s, int ld, int row0, int k0) {
  int l = threadIdx.x & 31, g = l >> 4;
  const _Float16* p = s + (row0 + (l & 15)) * (long)ld + k0 + g * 16;
  v8h lo = *(const v8h*)p;
  v8h hi = *(const v8h*)(p + 8);
  return __builtin_shufflevector(lo, hi, 0, 1, 2, 3, 4, 5, 6, 7, 8, 9, 10, 11, 12,
                                 13, 14, 15);
}

// B fragment (32x16 KxN) from K-contiguous ("transposed") storage:
// element (k, c) lives at bp[c*ld + k]. Per-lane one 32B vector load.
__device__ __forceinline__ v16h frag_bT_f16(const _Float16* bp, int ld, int k0, int c0) {
  int l = threadIdx.x & 31, g = l >> 4;
  return *(const v16h*)(bp + (long)(c0 + (l & 15)) * ld + k0 + g * 16);
}

// ---------------------------------------------------------------- small kernels
__global__ void k_zero(unsigned int* counts) {
  if (threadIdx.x < Nsz) counts[threadIdx.x] = 0u;
}

// Plain f32 -> f16 convert (layout preserved).
__global__ void k_cvt(const float* __restrict__ src, _Float16* __restrict__ dst,
                      long nelem) {
  long i = (long)blockIdx.x * 256 + threadIdx.x;
  if (i < nelem) dst[i] = (_Float16)src[i];
}

// Tiled transpose + convert: src [R][C] f32 (batch z) -> dst [C][R] f16.
__global__ void k_transpose_cvt(const float* __restrict__ src,
                                _Float16* __restrict__ dst, int R, int C) {
  long base = (long)blockIdx.z * R * C;
  int r0 = blockIdx.x * 32, c0 = blockIdx.y * 32;
  __shared__ float tile[32][33];
  int tid = threadIdx.x;
  int tr = tid >> 3, tc4 = (tid & 7) * 4;
#pragma unroll
  for (int j = 0; j < 4; ++j)
    tile[tr][tc4 + j] = src[base + (long)(r0 + tr) * C + c0 + tc4 + j];
  __syncthreads();
#pragma unroll
  for (int j = 0; j < 4; ++j)
    dst[base + (long)(c0 + tr) * R + r0 + tc4 + j] = (_Float16)tile[tc4 + j][tr];
}

// Router: one wave per token; top-k on logits (softmax is monotone).
__global__ void k_router(const float* __restrict__ x, const float* __restrict__ Wr,
                         const float* __restrict__ br, unsigned int* __restrict__ counts,
                         unsigned int* __restrict__ pairs) {
  int wave = threadIdx.x >> 5;
  int lane = threadIdx.x & 31;
  int t = blockIdx.x * 8 + wave;
  __shared__ float lg[8][32];
  const float* xr = x + (long)t * Dsz;
  const float* wr = Wr + (long)lane * Dsz;
  float acc = br[lane];
  for (int d = 0; d < Dsz; d += 4) {
    acc += xr[d] * wr[d] + xr[d + 1] * wr[d + 1] + xr[d + 2] * wr[d + 2] +
           xr[d + 3] * wr[d + 3];
  }
  lg[wave][lane] = acc;
  __syncthreads();
  if (lane == 0) {
    unsigned chosen = 0;
    for (int kk = 0; kk < Ksel; ++kk) {
      float best = -3.4e38f;
      int bi = 0;
      for (int n2 = 0; n2 < Nsz; ++n2)
        if (!((chosen >> n2) & 1u) && lg[wave][n2] > best) { best = lg[wave][n2]; bi = n2; }
      chosen |= 1u << bi;
      unsigned pos = atomicAdd(&counts[bi], 1u);
      pairs[bi * TOK + pos] = (unsigned)(t * Ksel + kk);
    }
  }
}

// Grouped FFN: block = (neuron n, 32-pair tile), 2 M-subtiles per wave.
// W1c: [N][H][D] f16 (K=D contiguous), W2c: [N][D][H] f16 (K=H contiguous).
#define LDA 520
#define LDH 136
__global__ void k_ffn(const float* __restrict__ x, const _Float16* __restrict__ W1c,
                      const float* __restrict__ b1, const _Float16* __restrict__ W2c,
                      const float* __restrict__ b2,
                      const unsigned int* __restrict__ counts,
                      const unsigned int* __restrict__ pairs,
                      float* __restrict__ sel) {
  int n = blockIdx.x >> 6;     // 64 tiles of 32 pairs
  int tile = blockIdx.x & 63;
  int cnt = (int)counts[n];
  if (tile * 32 >= cnt) return;
  int nvalid = min(32, cnt - tile * 32);

  __shared__ _Float16 As[32 * LDA];
  __shared__ _Float16 Hs[32 * LDH];
  __shared__ int toks[32];
  __shared__ int slots[32];

  int tid = threadIdx.x;
  if (tid < 32) {
    unsigned e = pairs[n * TOK + tile * 32 + ((tid < nvalid) ? tid : 0)];
    toks[tid] = (int)(e >> 2);
    slots[tid] = (int)(e & 3u);
  }
  __syncthreads();
  {  // stage 32 x-rows (f32 -> f16), 8 threads per row
    int row = tid >> 3, seg = (tid & 7) * 64;
    const float* src = x + (long)toks[row] * Dsz + seg;
    _Float16* dst = As + row * LDA + seg;
    for (int j = 0; j < 64; ++j) dst[j] = (_Float16)src[j];
  }
  __syncthreads();

  int w = tid >> 5, l = tid & 31, g = l >> 4;
  const _Float16* W1n = W1c + (long)n * Hsz * Dsz;  // [H][D]
  const _Float16* W2n = W2c + (long)n * Dsz * Hsz;  // [D][H]
  v8f c2[4][2] = {{{}, {}}, {{}, {}}, {{}, {}}, {{}, {}}};

  for (int hc = 0; hc < Hsz; hc += 128) {
    int hcol = hc + w * 16;
    v8f ch[2] = {{}, {}};
    for (int kk = 0; kk < Dsz; kk += 32) {
      v16h b = frag_bT_f16(W1n, Dsz, kk, hcol);
#pragma unroll
      for (int mt = 0; mt < 2; ++mt) {
        v16h a = frag_a(As, LDA, mt * 16, kk);
        ch[mt] = wmma16(a, b, ch[mt]);
      }
    }
    float bb = b1[n * Hsz + hcol + (l & 15)];
#pragma unroll
    for (int mt = 0; mt < 2; ++mt)
#pragma unroll
      for (int r = 0; r < 8; ++r) {  // bias + tanh-gelu, store f16
        float v = ch[mt][r] + bb;
        float gl = 0.5f * v * (1.0f + tanhf(0.7978845608f * (v + 0.044715f * v * v * v)));
        Hs[(mt * 16 + r + g * 8) * LDH + w * 16 + (l & 15)] = (_Float16)gl;
      }
    __syncthreads();
#pragma unroll
    for (int ti = 0; ti < 4; ++ti)
#pragma unroll
      for (int kk = 0; kk < 128; kk += 32) {
        v16h b = frag_bT_f16(W2n, Hsz, hc + kk, w * 64 + ti * 16);
#pragma unroll
        for (int mt = 0; mt < 2; ++mt) {
          v16h a = frag_a(Hs, LDH, mt * 16, kk);
          c2[ti][mt] = wmma16(a, b, c2[ti][mt]);
        }
      }
    __syncthreads();
  }
#pragma unroll
  for (int ti = 0; ti < 4; ++ti) {
    int co = w * 64 + ti * 16 + (l & 15);
    float bb = b2[n * Dsz + co];
#pragma unroll
    for (int mt = 0; mt < 2; ++mt)
#pragma unroll
      for (int r = 0; r < 8; ++r) {
        int m = mt * 16 + r + g * 8;
        if (m < nvalid)
          sel[((long)toks[m] * Ksel + slots[m]) * Dsz + co] = 0.25f * (c2[ti][mt][r] + bb);
      }
  }
}

// neuron_out = sum over K slots (already /K-scaled); emit f32 + f16 copies.
__global__ void k_reduce(const float* __restrict__ sel, float* __restrict__ nout,
                         _Float16* __restrict__ nout_h) {
  long i = (long)blockIdx.x * 256 + threadIdx.x;  // < TOK*Dsz
  long t = i >> 9;
  int d = (int)(i & 511);
  const float* s = sel + (t * Ksel) * Dsz + d;
  float v = s[0] + s[Dsz] + s[2 * Dsz] + s[3 * Dsz];
  nout[i] = v;
  nout_h[i] = (_Float16)v;
}

// QKV projection: [2048 x 512] @ Wqkv^T -> q/k ([bh][s][d]) and v transposed
// ([bh][d][s]) f16 buffers. Wqkvh is the f16 copy of Wqkv [1536(out)][512(in)],
// which is already K-contiguous for frag_bT.
#define LDQA 40
__global__ void k_qkv(const _Float16* __restrict__ Ah, const _Float16* __restrict__ Wqkvh,
                      const float* __restrict__ bias, _Float16* __restrict__ qb,
                      _Float16* __restrict__ kb, _Float16* __restrict__ vT) {
  int bm = blockIdx.x & 15;   // 16 row blocks of 128
  int bn = blockIdx.x >> 4;   // 24 col blocks of 64
  int tid = threadIdx.x, w = tid >> 5, l = tid & 31, g = l >> 4;
  __shared__ _Float16 As[128 * LDQA];
  v8f c[4] = {{}, {}, {}, {}};
  int row0 = bm * 128, col0 = bn * 64;

  for (int k0 = 0; k0 < Dsz; k0 += 32) {
    {  // stage 128x32 A chunk
      int r = tid >> 1, half = tid & 1;
      const _Float16* src = Ah + (long)(row0 + r) * Dsz + k0 + half * 16;
      _Float16* dst = As + r * LDQA + half * 16;
      *(v8h*)dst = *(const v8h*)src;
      *(v8h*)(dst + 8) = *(const v8h*)(src + 8);
    }
    __syncthreads();
    v16h a = frag_a(As, LDQA, w * 16, 0);
#pragma unroll
    for (int ti = 0; ti < 4; ++ti) {
      v16h b = frag_bT_f16(Wqkvh, Dsz, k0, col0 + ti * 16);
      c[ti] = wmma16(a, b, c[ti]);
    }
    __syncthreads();
  }
#pragma unroll
  for (int ti = 0; ti < 4; ++ti) {
    int co = col0 + ti * 16 + (l & 15);
    float bs = bias[co];
    int sect = co >> 9;  // 0=q 1=k 2=v (uniform per tile)
    int cq = co & 511, h = cq >> 6, d = cq & 63;
#pragma unroll
    for (int r = 0; r < 8; ++r) {
      int t = row0 + w * 16 + r + g * 8;
      int bb = t >> 10, s = t & 1023;
      _Float16 val = (_Float16)(c[ti][r] + bs);
      long bh = (long)(bb * NHsz + h);
      if (sect == 0)      qb[(bh * Ssz + s) * HDsz + d] = val;
      else if (sect == 1) kb[(bh * Ssz + s) * HDsz + d] = val;
      else                vT[(bh * HDsz + d) * Ssz + s] = val;
    }
  }
}

// Flash attention: per wave a 16-query tile over one (b,h); 32-key chunks.
// K stored [s][d] (K^T frag is contiguous); V stored [d][s] (PV frag contiguous).
#define LDP 40
__global__ void k_attn(const _Float16* __restrict__ qbuf, const _Float16* __restrict__ kbuf,
                       const _Float16* __restrict__ vT, _Float16* __restrict__ attn) {
  int qblk = blockIdx.x & 7;
  int bh = blockIdx.x >> 3;  // 0..15
  int bidx = bh >> 3, h = bh & 7;
  int tid = threadIdx.x, w = tid >> 5, l = tid & 31, g = l >> 4;
  __shared__ _Float16 pl[8][16 * LDP];

  const _Float16* qp = qbuf + (long)bh * Ssz * HDsz;
  const _Float16* kp = kbuf + (long)bh * Ssz * HDsz;
  const _Float16* vp = vT + (long)bh * HDsz * Ssz;
  int q0 = qblk * 128 + w * 16;

  v16h qa0 = frag_a(qp + (long)q0 * HDsz, HDsz, 0, 0);
  v16h qa1 = frag_a(qp + (long)q0 * HDsz, HDsz, 0, 32);
  v8f O[4] = {{}, {}, {}, {}};
  float mrow[8], lrow[8];
#pragma unroll
  for (int r = 0; r < 8; ++r) { mrow[r] = -3.0e38f; lrow[r] = 0.f; }

  for (int kk = 0; kk < Ssz; kk += 32) {
    v8f s0 = {}, s1 = {};
    s0 = wmma16(qa0, frag_bT_f16(kp, HDsz, 0, kk), s0);
    s0 = wmma16(qa1, frag_bT_f16(kp, HDsz, 32, kk), s0);
    s1 = wmma16(qa0, frag_bT_f16(kp, HDsz, 0, kk + 16), s1);
    s1 = wmma16(qa1, frag_bT_f16(kp, HDsz, 32, kk + 16), s1);
#pragma unroll
    for (int r = 0; r < 8; ++r) {
      s0[r] *= 0.125f;  // 1/sqrt(64)
      s1[r] *= 0.125f;
      float t = fmaxf(s0[r], s1[r]);
#pragma unroll
      for (int off = 1; off < 16; off <<= 1) t = fmaxf(t, __shfl_xor(t, off, 32));
      float nm = fmaxf(mrow[r], t);
      float corr = expf(mrow[r] - nm);
      mrow[r] = nm;
      float p0 = expf(s0[r] - nm), p1 = expf(s1[r] - nm);
      float rs = p0 + p1;
#pragma unroll
      for (int off = 1; off < 16; off <<= 1) rs += __shfl_xor(rs, off, 32);
      lrow[r] = lrow[r] * corr + rs;
#pragma unroll
      for (int ti = 0; ti < 4; ++ti) O[ti][r] *= corr;
      pl[w][(r + g * 8) * LDP + (l & 15)] = (_Float16)p0;
      pl[w][(r + g * 8) * LDP + 16 + (l & 15)] = (_Float16)p1;
    }
    v16h pa = frag_a(pl[w], LDP, 0, 0);  // 16x32 probs, C->A relayout via LDS
#pragma unroll
    for (int ti = 0; ti < 4; ++ti) {
      v16h b = frag_bT_f16(vp, Ssz, kk, ti * 16);
      O[ti] = wmma16(pa, b, O[ti]);
    }
  }
#pragma unroll
  for (int ti = 0; ti < 4; ++ti) {
#pragma unroll
    for (int r = 0; r < 8; ++r) {
      int srow = q0 + r + g * 8;
      float o = O[ti][r] / lrow[r];
      attn[((long)(bidx * Ssz + srow)) * Dsz + h * HDsz + ti * 16 + (l & 15)] = (_Float16)o;
    }
  }
}

// Fused: out-proj + residual + LN1 + residual + LN2. Woh = f16 Wo (K-contiguous).
__global__ void k_final(const _Float16* __restrict__ attn, const _Float16* __restrict__ Woh,
                        const float* __restrict__ bo, const float* __restrict__ x,
                        const float* __restrict__ nout, const float* __restrict__ ln1w,
                        const float* __restrict__ ln1b, const float* __restrict__ ln2w,
                        const float* __restrict__ ln2b, float* __restrict__ out) {
  __shared__ _Float16 As[16 * LDA];
  __shared__ float rowbuf[16][516];
  __shared__ float redS[16][17], redQ[16][17];
  __shared__ float mus[16], rss[16];
  int tid = threadIdx.x, w = tid >> 5, l = tid & 31, g = l >> 4;
  int row0 = blockIdx.x * 16;
  {  // stage 16 attn rows
    int r = tid >> 4, seg = tid & 15;
    const _Float16* src = attn + (long)(row0 + r) * Dsz + seg * 32;
    _Float16* dst = As + r * LDA + seg * 32;
    *(v8h*)dst = *(const v8h*)src;
    *(v8h*)(dst + 8) = *(const v8h*)(src + 8);
    *(v8h*)(dst + 16) = *(const v8h*)(src + 16);
    *(v8h*)(dst + 24) = *(const v8h*)(src + 24);
  }
  __syncthreads();
  v8f c[4] = {{}, {}, {}, {}};
  for (int k0 = 0; k0 < Dsz; k0 += 32) {
    v16h a = frag_a(As, LDA, 0, k0);
#pragma unroll
    for (int ti = 0; ti < 4; ++ti)
      c[ti] = wmma16(a, frag_bT_f16(Woh, Dsz, k0, w * 64 + ti * 16), c[ti]);
  }
#pragma unroll
  for (int ti = 0; ti < 4; ++ti) {
    int co = w * 64 + ti * 16 + (l & 15);
    float bs = bo[co];
#pragma unroll
    for (int r = 0; r < 8; ++r) {
      int m = r + g * 8;
      rowbuf[m][co] = c[ti][r] + bs + x[(long)(row0 + m) * Dsz + co];
    }
  }
  __syncthreads();
  int rowm = tid & 15, seg = tid >> 4;
  {  // LN1
    float s = 0.f, q = 0.f;
#pragma unroll
    for (int j = 0; j < 32; ++j) { float v = rowbuf[rowm][seg * 32 + j]; s += v; q += v * v; }
    redS[rowm][seg] = s; redQ[rowm][seg] = q;
  }
  __syncthreads();
  if (tid < 16) {
    float s = 0.f, q = 0.f;
#pragma unroll
    for (int j = 0; j < 16; ++j) { s += redS[tid][j]; q += redQ[tid][j]; }
    float mu = s * (1.f / 512.f);
    mus[tid] = mu;
    rss[tid] = rsqrtf(q * (1.f / 512.f) - mu * mu + 1e-5f);
  }
  __syncthreads();
  {
    float mu = mus[rowm], rs = rss[rowm];
#pragma unroll
    for (int j = 0; j < 32; ++j) {
      int ci = seg * 32 + j;
      float v = (rowbuf[rowm][ci] - mu) * rs * ln1w[ci] + ln1b[ci] +
                nout[(long)(row0 + rowm) * Dsz + ci];
      rowbuf[rowm][ci] = v;
    }
  }
  __syncthreads();
  {  // LN2
    float s = 0.f, q = 0.f;
#pragma unroll
    for (int j = 0; j < 32; ++j) { float v = rowbuf[rowm][seg * 32 + j]; s += v; q += v * v; }
    redS[rowm][seg] = s; redQ[rowm][seg] = q;
  }
  __syncthreads();
  if (tid < 16) {
    float s = 0.f, q = 0.f;
#pragma unroll
    for (int j = 0; j < 16; ++j) { s += redS[tid][j]; q += redQ[tid][j]; }
    float mu = s * (1.f / 512.f);
    mus[tid] = mu;
    rss[tid] = rsqrtf(q * (1.f / 512.f) - mu * mu + 1e-5f);
  }
  __syncthreads();
  {
    float mu = mus[rowm], rs = rss[rowm];
#pragma unroll
    for (int j = 0; j < 32; ++j) {
      int ci = seg * 32 + j;
      out[(long)(row0 + rowm) * Dsz + ci] =
          (rowbuf[rowm][ci] - mu) * rs * ln2w[ci] + ln2b[ci];
    }
  }
}

// ---------------------------------------------------------------- launch
extern "C" void kernel_launch(void* const* d_in, const int* in_sizes, int n_in,
                              void* d_out, int out_size, void* d_ws, size_t ws_size,
                              hipStream_t stream) {
  const float* x    = (const float*)d_in[0];
  const float* Wr   = (const float*)d_in[1];
  const float* br   = (const float*)d_in[2];
  const float* W1   = (const float*)d_in[3];
  const float* b1   = (const float*)d_in[4];
  const float* W2   = (const float*)d_in[5];
  const float* b2   = (const float*)d_in[6];
  const float* Wqkv = (const float*)d_in[7];
  const float* bqkv = (const float*)d_in[8];
  const float* Wo   = (const float*)d_in[9];
  const float* bo   = (const float*)d_in[10];
  const float* ln1w = (const float*)d_in[11];
  const float* ln1b = (const float*)d_in[12];
  const float* ln2w = (const float*)d_in[13];
  const float* ln2b = (const float*)d_in[14];
  float* out = (float*)d_out;

  char* p = (char*)d_ws;
  auto alloc = [&](size_t bytes) -> void* {
    void* r = (void*)p;
    p += (bytes + 255) & ~(size_t)255;
    return r;
  };
  unsigned* counts  = (unsigned*)alloc(Nsz * sizeof(unsigned));
  unsigned* pairs   = (unsigned*)alloc((size_t)Nsz * TOK * sizeof(unsigned));
  float* sel        = (float*)alloc((size_t)TOK * Ksel * Dsz * sizeof(float));
  float* nout       = (float*)alloc((size_t)TOK * Dsz * sizeof(float));
  _Float16* nout_h  = (_Float16*)alloc((size_t)TOK * Dsz * sizeof(_Float16));
  _Float16* W1c     = (_Float16*)alloc((size_t)Nsz * Hsz * Dsz * sizeof(_Float16));
  _Float16* W2c     = (_Float16*)alloc((size_t)Nsz * Dsz * Hsz * sizeof(_Float16));
  _Float16* Wqkvh   = (_Float16*)alloc((size_t)3 * Dsz * Dsz * sizeof(_Float16));
  _Float16* Woh     = (_Float16*)alloc((size_t)Dsz * Dsz * sizeof(_Float16));
  _Float16* qbuf    = (_Float16*)alloc((size_t)Bsz * NHsz * Ssz * HDsz * sizeof(_Float16));
  _Float16* kbuf    = (_Float16*)alloc((size_t)Bsz * NHsz * Ssz * HDsz * sizeof(_Float16));
  _Float16* vTb     = (_Float16*)alloc((size_t)Bsz * NHsz * Ssz * HDsz * sizeof(_Float16));
  _Float16* attnb   = (_Float16*)alloc((size_t)TOK * Dsz * sizeof(_Float16));

  k_zero<<<1, 32, 0, stream>>>(counts);
  k_router<<<TOK / 8, 256, 0, stream>>>(x, Wr, br, counts, pairs);
  k_cvt<<<((long)3 * Dsz * Dsz + 255) / 256, 256, 0, stream>>>(Wqkv, Wqkvh,
                                                               (long)3 * Dsz * Dsz);
  k_cvt<<<((long)Dsz * Dsz + 255) / 256, 256, 0, stream>>>(Wo, Woh, (long)Dsz * Dsz);
  {  // W1 [N][D][H] -> W1c [N][H][D]
    dim3 g1(Dsz / 32, Hsz / 32, Nsz);
    k_transpose_cvt<<<g1, 256, 0, stream>>>(W1, W1c, Dsz, Hsz);
  }
  {  // W2 [N][H][D] -> W2c [N][D][H]
    dim3 g2(Hsz / 32, Dsz / 32, Nsz);
    k_transpose_cvt<<<g2, 256, 0, stream>>>(W2, W2c, Hsz, Dsz);
  }
  k_ffn<<<Nsz * 64, 256, 0, stream>>>(x, W1c, b1, W2c, b2, counts, pairs, sel);
  k_reduce<<<(TOK * Dsz) / 256, 256, 0, stream>>>(sel, nout, nout_h);
  k_qkv<<<16 * 24, 256, 0, stream>>>(nout_h, Wqkvh, bqkv, qbuf, kbuf, vTb);
  k_attn<<<Bsz * NHsz * (Ssz / 128), 256, 0, stream>>>(qbuf, kbuf, vTb, attnb);
  k_final<<<TOK / 16, 256, 0, stream>>>(attnb, Woh, bo, x, nout, ln1w, ln1b, ln2w,
                                        ln2b, out);
}
